// QAdaptHypergraphConv_65463891526212
// MI455X (gfx1250) — matrix operand order repaired and checked by reading, not compiled, not run on hardware
//
#include <hip/hip_runtime.h>
#include <hip/hip_bf16.h>
#include <math.h>

typedef _Float16 half_t;
typedef __attribute__((ext_vector_type(8)))  _Float16 v8h;
typedef __attribute__((ext_vector_type(16))) _Float16 v16h;
typedef __attribute__((ext_vector_type(8)))  float    v8f;
typedef __attribute__((ext_vector_type(4)))  int      v4i_t;

#define Bc 4
#define Nn 4096
#define Ee 256
#define Ff 256
#define MAXE 50
#define H1c 128
#define H2c 64
#define BN (Bc*Nn)   // 16384

// ---------- workspace byte offsets ----------
#define OFF_XH      ((size_t)0)                    // BN*256 f16 = 8 MB
#define OFF_XP      ((size_t)8388608)              // 8 MB
#define OFF_XW      ((size_t)16777216)             // 8 MB (stored transposed [b][f][n])
#define OFF_WNT     ((size_t)25165824)             // 256*256 f16 = 128 KB
#define OFF_WT      ((size_t)25296896)             // 128 KB
#define OFF_COMPAT  ((size_t)25427968)             // BN f32 = 64 KB
#define OFF_MEAN    ((size_t)25493504)             // B*50*256 f32 = 200 KB
#define OFF_CTX     ((size_t)25698304)             // B*50 f32
#define OFF_DEG     ((size_t)25699328)             // 50 f32

// ---------- CDNA5 async global->LDS copy (guarded) ----------
#if defined(__has_builtin)
# if __has_builtin(__builtin_amdgcn_global_load_async_to_lds_b128)
#  define USE_ASYNC_LDS 1
# endif
#endif
#ifndef USE_ASYNC_LDS
# define USE_ASYNC_LDS 0
#endif

__device__ inline void cp16_g2l(const half_t* g, half_t* l) {
#if USE_ASYNC_LDS
  __builtin_amdgcn_global_load_async_to_lds_b128(
      (__attribute__((address_space(1))) v4i_t*)g,
      (__attribute__((address_space(3))) v4i_t*)l, 0, 0);
#else
  *(v8h*)l = *(const v8h*)g;
#endif
}

__device__ inline void wait_async0() {
#if defined(__AMDGCN__)
  asm volatile("s_wait_asynccnt 0x0" ::: "memory");
#endif
}
__device__ inline void wait_ds0() {
#if defined(__AMDGCN__)
  asm volatile("s_wait_dscnt 0x0" ::: "memory");
#endif
}

// ---------- helpers ----------
__device__ inline v16h ldv16(const half_t* p0, const half_t* p1) {
  union { v16h v; v8h h[2]; } u;
  u.h[0] = *(const v8h*)p0;
  u.h[1] = *(const v8h*)p1;
  return u.v;
}
__device__ inline v16h ldv16c(const half_t* p) { return ldv16(p, p + 8); }

__device__ inline v8f wmma_f16(v16h a, v16h b, v8f c) {
  return __builtin_amdgcn_wmma_f32_16x16x32_f16(false, a, false, b, (short)0, c,
                                                false, false);
}

// block-wide sum reduce, 256 threads
__device__ inline float blk_reduce256(float v, float* red) {
  int t = threadIdx.x;
  red[t] = v; __syncthreads();
  #pragma unroll
  for (int s = 128; s > 0; s >>= 1) {
    if (t < s) red[t] += red[t + s];
    __syncthreads();
  }
  float r = red[0]; __syncthreads();
  return r;
}

// ---------- kernel 1: x -> f16, and compat = x @ c_w + c_b ----------
__global__ __launch_bounds__(256) void k_convert_x(const float* __restrict__ x,
                                                   const float* __restrict__ c_w,
                                                   const float* __restrict__ c_b,
                                                   half_t* __restrict__ xh,
                                                   float* __restrict__ compat) {
  __shared__ float red[256];
  int row = blockIdx.x, t = threadIdx.x;
  float v = x[(size_t)row * Ff + t];
  xh[(size_t)row * Ff + t] = (half_t)v;
  float s = blk_reduce256(v * c_w[t], red);
  if (t == 0) compat[row] = s + c_b[0];
}

// ---------- kernel 2: W[k][n] f32 -> WT[n][k] f16 ----------
__global__ __launch_bounds__(256) void k_convert_wT(const float* __restrict__ W,
                                                    half_t* __restrict__ WT) {
  int idx = blockIdx.x * 256 + threadIdx.x;   // 65536 elems
  int k = idx >> 8, n = idx & 255;
  WT[n * 256 + k] = (half_t)W[idx];
}

// ---------- kernel 3: C_f16 = A_f16 @ W (WT f16 given), +bias ----------
// store_t == 0: C[row][col]   (row-major [16384][256])
// store_t == 1: C[b][col][n]  (per-batch transposed [4][256][4096])
__global__ __launch_bounds__(256) void k_gemm(const half_t* __restrict__ A,
                                              const half_t* __restrict__ BT,
                                              const float* __restrict__ bias,
                                              int add_bias, int store_t,
                                              half_t* __restrict__ C) {
  int lane = threadIdx.x & 31, wv = threadIdx.x >> 5;
  int tid = blockIdx.x * 8 + wv;       // 16384 wave-tiles
  int mt = tid >> 4, nt = tid & 15;
  int m = lane & 15, hi = lane >> 4;
  const half_t* arow = A + ((size_t)(mt * 16 + m)) * 256;
  const half_t* brow = BT + ((size_t)(nt * 16 + m)) * 256;
  v8f c = {};
  #pragma unroll
  for (int f0 = 0; f0 < 8; ++f0) {
    v16h a = ldv16c(arow + f0 * 32 + hi * 16);
    v16h b = ldv16c(brow + f0 * 32 + hi * 16);
    c = wmma_f16(a, b, c);
  }
  float bv = add_bias ? bias[nt * 16 + m] : 0.0f;
  int col = nt * 16 + m;
  #pragma unroll
  for (int j = 0; j < 8; ++j) {
    int row = mt * 16 + j + hi * 8;
    half_t val = (half_t)(c[j] + bv);
    if (!store_t) {
      C[(size_t)row * 256 + col] = val;
    } else {
      int bb = row >> 12, n = row & 4095;
      C[((size_t)(bb * 256 + col)) * 4096 + n] = val;
    }
  }
}

// ---------- kernel 4: per-edge degree + mean features ----------
__global__ __launch_bounds__(256) void k_edge_mean(const float* __restrict__ x,
                                                   const int* __restrict__ H,
                                                   float* __restrict__ mean,
                                                   float* __restrict__ deg_out) {
  __shared__ float red[256];
  int e = blockIdx.x, b = blockIdx.y, t = threadIdx.x;
  float d = 0.f;
  for (int n = t; n < Nn; n += 256) d += (H[n * Ee + e] != 0) ? 1.f : 0.f;
  float deg = blk_reduce256(d, red);
  if (t == 0 && b == 0) deg_out[e] = deg;
  float acc = 0.f;
  for (int n = 0; n < Nn; ++n) {
    if (H[n * Ee + e] != 0)                       // uniform branch: skips loads
      acc += x[((size_t)b * Nn + n) * Ff + t];
  }
  mean[((size_t)b * MAXE + e) * Ff + t] = acc / fmaxf(deg, 1.0f);
}

// ---------- kernel 5: MLP_context 256->128->64->1 ----------
__global__ __launch_bounds__(128) void k_mlp_ctx(const float* __restrict__ mean,
                                                 const float* __restrict__ m1w, const float* __restrict__ m1b,
                                                 const float* __restrict__ m2w, const float* __restrict__ m2b,
                                                 const float* __restrict__ m3w, const float* __restrict__ m3b,
                                                 float* __restrict__ ctx) {
  __shared__ float sm[256], h1[128], h2[64], red[64];
  int id = blockIdx.x, t = threadIdx.x;          // id = b*50 + e
  const float* mrow = mean + (size_t)id * 256;
  sm[t] = mrow[t]; sm[t + 128] = mrow[t + 128];
  __syncthreads();
  float acc = m1b[t];
  for (int k = 0; k < 256; ++k) acc += sm[k] * m1w[k * H1c + t];
  h1[t] = fmaxf(acc, 0.f);
  __syncthreads();
  if (t < 64) {
    float a2 = m2b[t];
    for (int k = 0; k < 128; ++k) a2 += h1[k] * m2w[k * H2c + t];
    h2[t] = fmaxf(a2, 0.f);
  }
  __syncthreads();
  if (t < 64) red[t] = h2[t] * m3w[t];
  __syncthreads();
  #pragma unroll
  for (int s = 32; s > 0; s >>= 1) {
    if (t < s) red[t] += red[t + s];
    __syncthreads();
  }
  if (t == 0) ctx[id] = red[0] + m3b[0];
}

// ---------- kernel 6: adaptive weights -> out[:, :256] ----------
__global__ __launch_bounds__(256) void k_gamma(const int* __restrict__ H,
                                               const float* __restrict__ compat,
                                               const float* __restrict__ ctx,
                                               const float* __restrict__ deg,
                                               const float* __restrict__ hedge_bias,
                                               const float* __restrict__ alpha,
                                               float* __restrict__ out) {
  int row = blockIdx.x;                  // b*4096 + n
  int b = row >> 12, n = row & 4095, e = threadIdx.x;
  float val = 0.f;
  if (e < MAXE) {
    if (H[n * Ee + e] != 0 && deg[e] > 1.0f) {
      float z = compat[row] + hedge_bias[0] + alpha[0] * ctx[b * MAXE + e];
      val = 1.0f / (1.0f + __expf(-z));
    }
  }
  out[(size_t)row * 512 + e] = val;
}

// ---------- kernel 7: flash attention -> out[:, 256:512] ----------
// xp: [b][n][f] f16 row-major; xwT: [b][f][n] f16 (pre-transposed V)
__global__ __launch_bounds__(256) void k_flash(const half_t* __restrict__ xp,
                                               const half_t* __restrict__ xwT,
                                               const float* __restrict__ bias,
                                               float* __restrict__ out) {
  __shared__ __align__(16) half_t sK[2][32 * 256];   // keys' xp rows (double buffer)
  __shared__ __align__(16) half_t sVT[2][256 * 32];  // xw transposed [f][k] (double buffer)
  __shared__ __align__(16) half_t sP[8 * 16 * 32];   // per-wave P scratch
  int lane = threadIdx.x & 31, wv = threadIdx.x >> 5;
  int b = blockIdx.y;
  int qb = blockIdx.x * 128 + wv * 16;
  int m = lane & 15, hi = lane >> 4;

  const half_t* kroot = xp + ((size_t)b * Nn) * 256;
  const half_t* vroot = xwT + ((size_t)b * 256) * 4096;

  // preload Q fragments (16 rows x 256) as 8 A-fragments
  const half_t* qrow = xp + ((size_t)(b * Nn + qb + m)) * 256;
  v16h qf[8];
  #pragma unroll
  for (int f0 = 0; f0 < 8; ++f0)
    qf[f0] = ldv16c(qrow + f0 * 32 + hi * 16);

  v8f o[16];
  #pragma unroll
  for (int ft = 0; ft < 16; ++ft) o[ft] = (v8f){};
  float mrow[8], lrow[8];
  #pragma unroll
  for (int j = 0; j < 8; ++j) { mrow[j] = -1e30f; lrow[j] = 0.f; }

  // stage one 32-key block (K rows row-major, V columns contiguous from xwT)
  auto stage = [&](int buf, int k0) {
    const half_t* kb = kroot + (size_t)k0 * 256;
    for (int i = threadIdx.x; i < 1024; i += 256)
      cp16_g2l(kb + i * 8, &sK[buf][i * 8]);
    for (int i = threadIdx.x; i < 1024; i += 256) {
      int f = i >> 2, p = i & 3;
      cp16_g2l(vroot + (size_t)f * 4096 + k0 + p * 8, &sVT[buf][f * 32 + p * 8]);
    }
  };

  stage(0, 0);   // prologue prefetch

  for (int kb = 0; kb < Nn / 32; ++kb) {
    int cur = kb & 1;
    wait_async0();
    __syncthreads();
    if (kb + 1 < Nn / 32) stage(cur ^ 1, (kb + 1) * 32);  // async prefetch next

    // scores: two 16x16 tiles (keys +0..15, +16..31)
    v8f s0 = {}, s1 = {};
    #pragma unroll
    for (int f0 = 0; f0 < 8; ++f0) {
      v16h kf0 = ldv16c(&sK[cur][(m)      * 256 + f0 * 32 + hi * 16]);
      v16h kf1 = ldv16c(&sK[cur][(16 + m) * 256 + f0 * 32 + hi * 16]);
      s0 = wmma_f16(qf[f0], kf0, s0);
      s1 = wmma_f16(qf[f0], kf1, s1);
    }
    const float scale = 0.0625f;   // 1/sqrt(256)
    s0 = s0 * scale; s1 = s1 * scale;

    // online softmax across the 16-lane row groups
    float alpha[8];
    #pragma unroll
    for (int j = 0; j < 8; ++j) {
      float rv = fmaxf(s0[j], s1[j]);
      rv = fmaxf(rv, __shfl_xor(rv, 1, 32));
      rv = fmaxf(rv, __shfl_xor(rv, 2, 32));
      rv = fmaxf(rv, __shfl_xor(rv, 4, 32));
      rv = fmaxf(rv, __shfl_xor(rv, 8, 32));
      float nm = fmaxf(mrow[j], rv);
      alpha[j] = __expf(mrow[j] - nm);
      mrow[j] = nm;
      float p0 = __expf(s0[j] - nm);
      float p1 = __expf(s1[j] - nm);
      s0[j] = p0; s1[j] = p1;
      float rs = p0 + p1;
      rs += __shfl_xor(rs, 1, 32);
      rs += __shfl_xor(rs, 2, 32);
      rs += __shfl_xor(rs, 4, 32);
      rs += __shfl_xor(rs, 8, 32);
      lrow[j] = lrow[j] * alpha[j] + rs;
    }
    #pragma unroll
    for (int ft = 0; ft < 16; ++ft)
      #pragma unroll
      for (int j = 0; j < 8; ++j) o[ft][j] *= alpha[j];

    // re-layout P (C-layout) -> A-layout via per-wave LDS round trip
    half_t* pw = sP + wv * 512;
    #pragma unroll
    for (int j = 0; j < 8; ++j) {
      int q = j + hi * 8;
      pw[q * 32 + m]      = (half_t)s0[j];
      pw[q * 32 + 16 + m] = (half_t)s1[j];
    }
    wait_ds0();
    v16h pf = ldv16c(pw + m * 32 + hi * 16);

    // O += P @ V  (K = 32 keys, 16 feature tiles)
    #pragma unroll
    for (int ft = 0; ft < 16; ++ft) {
      v16h bv = ldv16c(&sVT[cur][(ft * 16 + m) * 32 + hi * 16]);
      o[ft] = wmma_f16(pf, bv, o[ft]);
    }
  }

  // epilogue: normalize, add bias, store out[:, 256:512]
  float rcp[8];
  #pragma unroll
  for (int j = 0; j < 8; ++j) rcp[j] = 1.0f / lrow[j];
  #pragma unroll
  for (int ft = 0; ft < 16; ++ft) {
    int f = ft * 16 + m;
    float bvv = bias[f];
    #pragma unroll
    for (int j = 0; j < 8; ++j) {
      int row = qb + j + hi * 8;
      out[((size_t)(b * Nn + row)) * 512 + 256 + f] = o[ft][j] * rcp[j] + bvv;
    }
  }
}

// ---------- host side ----------
extern "C" void kernel_launch(void* const* d_in, const int* in_sizes, int n_in,
                              void* d_out, int out_size, void* d_ws, size_t ws_size,
                              hipStream_t stream) {
  const float* x          = (const float*)d_in[0];
  const int*   H          = (const int*)  d_in[1];
  const float* weight     = (const float*)d_in[2];
  const float* bias       = (const float*)d_in[3];
  const float* Wn_w       = (const float*)d_in[4];
  const float* Wn_b       = (const float*)d_in[5];
  const float* m1w        = (const float*)d_in[6];
  const float* m1b        = (const float*)d_in[7];
  const float* m2w        = (const float*)d_in[8];
  const float* m2b        = (const float*)d_in[9];
  const float* m3w        = (const float*)d_in[10];
  const float* m3b        = (const float*)d_in[11];
  const float* c_w        = (const float*)d_in[12];
  const float* c_b        = (const float*)d_in[13];
  const float* hedge_bias = (const float*)d_in[14];
  const float* alpha_ctx  = (const float*)d_in[15];
  float* out = (float*)d_out;

  char* ws = (char*)d_ws;
  half_t* xh     = (half_t*)(ws + OFF_XH);
  half_t* xph    = (half_t*)(ws + OFF_XP);
  half_t* xwT    = (half_t*)(ws + OFF_XW);
  half_t* wnT    = (half_t*)(ws + OFF_WNT);
  half_t* wT     = (half_t*)(ws + OFF_WT);
  float*  compat = (float*)(ws + OFF_COMPAT);
  float*  meanE  = (float*)(ws + OFF_MEAN);
  float*  ctx    = (float*)(ws + OFF_CTX);
  float*  deg    = (float*)(ws + OFF_DEG);

  k_convert_x<<<BN, 256, 0, stream>>>(x, c_w, c_b, xh, compat);
  k_convert_wT<<<256, 256, 0, stream>>>(Wn_w, wnT);
  k_convert_wT<<<256, 256, 0, stream>>>(weight, wT);
  k_gemm<<<2048, 256, 0, stream>>>(xh, wnT, Wn_b, 1, 0, xph);   // xp, row-major
  k_gemm<<<2048, 256, 0, stream>>>(xh, wT,  Wn_b, 0, 1, xwT);   // xw, transposed
  k_edge_mean<<<dim3(MAXE, Bc), 256, 0, stream>>>(x, H, meanE, deg);
  k_mlp_ctx<<<Bc * MAXE, 128, 0, stream>>>(meanE, m1w, m1b, m2w, m2b, m3w, m3b, ctx);
  k_gamma<<<BN, 256, 0, stream>>>(H, compat, ctx, deg, hedge_bias, alpha_ctx, out);
  k_flash<<<dim3(32, Bc), 256, 0, stream>>>(xph, xwT, bias, out);
}
